// HydraMambaReg_34265249088354
// MI455X (gfx1250) — compile-verified, matrix operand order checked by reading, not run on hardware
//
#include <hip/hip_runtime.h>
#include <math.h>

// ---------------- model constants ----------------
#define BN_   4
#define NPTS_ 8192
#define GC_   1024
#define KN_   32
#define C0_   96
#define C1_   192
#define HEADS_ 4
#define DSTATE_ 16
#define DI_   768          // EXPAND*C1
#define DH_   192          // DI/HEADS
#define DTRANK_ 12
#define PROJW_ 140         // DTRANK + 2*HEADS*DSTATE
#define MLPH_ 768
#define NCLS_ 6
#define EPS_  1e-5f

#define ACT_NONE 0
#define ACT_RELU 1
#define ACT_SOFTPLUS 3

typedef __attribute__((ext_vector_type(16))) _Float16 v16h;
typedef __attribute__((ext_vector_type(8)))  _Float16 v8h;
typedef __attribute__((ext_vector_type(8)))  float    v8f;

// ---------------- device helpers ----------------
__device__ __forceinline__ float sigmoidf_(float x) { return 1.f / (1.f + __expf(-x)); }
__device__ __forceinline__ float siluf_(float x) { return x * sigmoidf_(x); }
__device__ __forceinline__ float geluf_(float x) {
  const float c = 0.7978845608028654f;
  return 0.5f * x * (1.f + tanhf(c * (x + 0.044715f * x * x * x)));
}
__device__ __forceinline__ float softplusf_(float x) {
  return (x > 20.f) ? x : log1pf(__expf(x));
}
__device__ __forceinline__ float epiloguef_(float v, const float* bias, int c, int act) {
  if (bias) v += bias[c];
  if (act == ACT_RELU) v = fmaxf(v, 0.f);
  else if (act == ACT_SOFTPLUS) v = softplusf_(v);
  return v;
}

// ---------------- fast WMMA GEMM ----------------
// Requires M%128==0, N%32==0, K%32==0, 16B-aligned A/W rows (lda%4==0, ldw%4==0).
// Block = 256 threads = 8 waves stacked in M. Block tile 128(M) x 32(N).
// Each wave: one 16x32 A fragment, two 16x16 WMMAs per K-step (cols 0-15, 16-31).
// B tile (32x32) staged to LDS as f16 in [col][k] layout -> ds_load_b128 fragments.
__global__ void gemm_f16_wmma_fast(const float* __restrict__ A, int lda,
                                   const float* __restrict__ W, int ldw,
                                   const float* __restrict__ bias,
                                   float* __restrict__ C, int ldc,
                                   int M, int N, int K, int act, int accum) {
  __shared__ _Float16 sB[32 * 32];     // [col][k]
  const int lane = threadIdx.x & 31;
  const int wid  = threadIdx.x >> 5;
  const int row0 = blockIdx.x * 128 + wid * 16;
  const int col0 = blockIdx.y * 32;
  const int half = lane >> 4;          // lanes 16-31 carry K+8 halves
  const int l15  = lane & 15;
  const int kb   = half * 8;
  const int arow = row0 + l15;
  // cooperative B staging: thread t loads W[k0 + t/8][col0 + (t%8)*4 .. +3]
  const int bk = threadIdx.x >> 3;
  const int bc = (threadIdx.x & 7) * 4;

  v8f acc0 = {}, acc1 = {};
  for (int k0 = 0; k0 < K; k0 += 32) {
    const float4 wv = *reinterpret_cast<const float4*>(
        W + (long long)(k0 + bk) * ldw + col0 + bc);
    __syncthreads();                   // prev-iter LDS reads done before overwrite
    sB[(bc + 0) * 32 + bk] = (_Float16)wv.x;
    sB[(bc + 1) * 32 + bk] = (_Float16)wv.y;
    sB[(bc + 2) * 32 + bk] = (_Float16)wv.z;
    sB[(bc + 3) * 32 + bk] = (_Float16)wv.w;

    // A fragment: two contiguous K-runs per lane -> 4x b128 loads
    const float* ap = A + (long long)arow * lda + k0 + kb;
    const float4 a01 = *reinterpret_cast<const float4*>(ap);
    const float4 a23 = *reinterpret_cast<const float4*>(ap + 4);
    const float4 a45 = *reinterpret_cast<const float4*>(ap + 16);
    const float4 a67 = *reinterpret_cast<const float4*>(ap + 20);
    if (k0 + 32 < K) __builtin_prefetch(ap + 32, 0, 1);
    v16h a;
    a[0]  = (_Float16)a01.x; a[1]  = (_Float16)a01.y;
    a[2]  = (_Float16)a01.z; a[3]  = (_Float16)a01.w;
    a[4]  = (_Float16)a23.x; a[5]  = (_Float16)a23.y;
    a[6]  = (_Float16)a23.z; a[7]  = (_Float16)a23.w;
    a[8]  = (_Float16)a45.x; a[9]  = (_Float16)a45.y;
    a[10] = (_Float16)a45.z; a[11] = (_Float16)a45.w;
    a[12] = (_Float16)a67.x; a[13] = (_Float16)a67.y;
    a[14] = (_Float16)a67.z; a[15] = (_Float16)a67.w;

    __syncthreads();                   // B tile visible to all waves
    // B fragments: lane's column, K-runs kb..kb+7 and kb+16..kb+23 (16B aligned)
    const _Float16* bp0 = &sB[l15 * 32 + kb];
    const _Float16* bp1 = &sB[(16 + l15) * 32 + kb];
    const v8h b0lo = *reinterpret_cast<const v8h*>(bp0);
    const v8h b0hi = *reinterpret_cast<const v8h*>(bp0 + 16);
    const v8h b1lo = *reinterpret_cast<const v8h*>(bp1);
    const v8h b1hi = *reinterpret_cast<const v8h*>(bp1 + 16);
    const v16h b0 = __builtin_shufflevector(b0lo, b0hi, 0, 1, 2, 3, 4, 5, 6, 7,
                                            8, 9, 10, 11, 12, 13, 14, 15);
    const v16h b1 = __builtin_shufflevector(b1lo, b1hi, 0, 1, 2, 3, 4, 5, 6, 7,
                                            8, 9, 10, 11, 12, 13, 14, 15);
    acc0 = __builtin_amdgcn_wmma_f32_16x16x32_f16(false, a, false, b0,
                                                  (short)0, acc0, false, false);
    acc1 = __builtin_amdgcn_wmma_f32_16x16x32_f16(false, a, false, b1,
                                                  (short)0, acc1, false, false);
  }

  const int c0 = col0 + l15;
  const int c1 = col0 + 16 + l15;
#pragma unroll
  for (int j = 0; j < 8; ++j) {
    const int r = row0 + j + half * 8;
    const float v0 = epiloguef_(acc0[j], bias, c0, act);
    const float v1 = epiloguef_(acc1[j], bias, c1, act);
    const long long o0 = (long long)r * ldc + c0;
    const long long o1 = (long long)r * ldc + c1;
    if (accum) { C[o0] += v0; C[o1] += v1; }
    else       { C[o0]  = v0; C[o1]  = v1; }
  }
}

// ---------------- generic bounds-checked WMMA GEMM (odd shapes only) ----------------
__global__ void gemm_f16_wmma(const float* __restrict__ A, int lda,
                              const float* __restrict__ W, int ldw,
                              const float* __restrict__ bias,
                              float* __restrict__ C, int ldc,
                              int M, int N, int K, int act, int accum) {
  const int lane = threadIdx.x & 31;
  const int wid  = threadIdx.x >> 5;
  const int row0 = blockIdx.x * 32 + (wid >> 2) * 16;
  const int col0 = blockIdx.y * 64 + (wid & 3) * 16;
  const int half = lane >> 4;
  const int l15  = lane & 15;
  const int arow = row0 + l15;
  const int bcol = col0 + l15;
  const int kb   = half * 8;

  v8f acc = {};
  for (int k0 = 0; k0 < K; k0 += 32) {
    v16h a = {}, b = {};
#pragma unroll
    for (int v = 0; v < 8; ++v) {
      const int kk = k0 + kb + ((v < 4) ? (2 * v) : (16 + 2 * (v - 4)));
      float a0 = 0.f, a1 = 0.f, b0 = 0.f, b1 = 0.f;
      if (arow < M) {
        if (kk     < K) a0 = A[(long long)arow * lda + kk];
        if (kk + 1 < K) a1 = A[(long long)arow * lda + kk + 1];
      }
      if (bcol < N) {
        if (kk     < K) b0 = W[(long long)kk * ldw + bcol];
        if (kk + 1 < K) b1 = W[(long long)(kk + 1) * ldw + bcol];
      }
      a[2 * v]     = (_Float16)a0;
      a[2 * v + 1] = (_Float16)a1;
      b[2 * v]     = (_Float16)b0;
      b[2 * v + 1] = (_Float16)b1;
    }
    acc = __builtin_amdgcn_wmma_f32_16x16x32_f16(false, a, false, b,
                                                 (short)0, acc, false, false);
  }
#pragma unroll
  for (int j = 0; j < 8; ++j) {
    const int r = row0 + j + half * 8;
    const int c = col0 + l15;
    if (r < M && c < N) {
      const float v = epiloguef_(acc[j], bias, c, act);
      const long long off = (long long)r * ldc + c;
      if (accum) C[off] += v; else C[off] = v;
    }
  }
}

// ---------------- FPS: one block per batch ----------------
__global__ void fps_kernel(const float* __restrict__ xyz, int* __restrict__ idx_out) {
  __shared__ float dists[NPTS_];
  __shared__ float rd[256];
  __shared__ int   ri[256];
  const int b = blockIdx.x, t = threadIdx.x;
  const float* p = xyz + (long long)b * NPTS_ * 3;
  for (int i = t; i < NPTS_; i += 256) dists[i] = 1e10f;
  int last = 0;
  if (t == 0) idx_out[b * GC_] = 0;
  __syncthreads();
  for (int it = 1; it < GC_; ++it) {
    const float lx = p[last * 3], ly = p[last * 3 + 1], lz = p[last * 3 + 2];
    float best = -1.f; int besti = 0;
    for (int i = t; i < NPTS_; i += 256) {
      const float dx = p[i * 3] - lx, dy = p[i * 3 + 1] - ly, dz = p[i * 3 + 2] - lz;
      const float m = fminf(dists[i], dx * dx + dy * dy + dz * dz);
      dists[i] = m;
      if (m > best) { best = m; besti = i; }
    }
    rd[t] = best; ri[t] = besti;
    __syncthreads();
    for (int s = 128; s > 0; s >>= 1) {
      if (t < s) {
        if (rd[t + s] > rd[t] || (rd[t + s] == rd[t] && ri[t + s] < ri[t])) {
          rd[t] = rd[t + s]; ri[t] = ri[t + s];
        }
      }
      __syncthreads();
    }
    last = ri[0];
    if (t == 0) idx_out[b * GC_ + it] = last;
    __syncthreads();
  }
}

__global__ void gather_center_kernel(const float* __restrict__ xyz,
                                     const int* __restrict__ fidx,
                                     float* __restrict__ center) {
  const int i = blockIdx.x * blockDim.x + threadIdx.x;
  if (i >= BN_ * GC_) return;
  const int b = i / GC_;
  const int src = fidx[i];
  const float* s = xyz + ((long long)b * NPTS_ + src) * 3;
  center[i * 3] = s[0]; center[i * 3 + 1] = s[1]; center[i * 3 + 2] = s[2];
}

// ---------------- KNN: iterative top-32-min, one block per center ----------------
__global__ void knn_kernel(const float* __restrict__ xyz,
                           const float* __restrict__ center,
                           int* __restrict__ knn_idx) {
  __shared__ float rd[256];
  __shared__ int   ri[256];
  __shared__ float s_ld; __shared__ int s_li;
  const int bg = blockIdx.x, t = threadIdx.x;
  const int b = bg / GC_;
  const float cx = center[bg * 3], cy = center[bg * 3 + 1], cz = center[bg * 3 + 2];
  const float* p = xyz + (long long)b * NPTS_ * 3;
  float lastd = -1e30f; int lasti = -1;
  for (int j = 0; j < KN_; ++j) {
    float best = 1e30f; int besti = 0x7fffffff;
    for (int i = t; i < NPTS_; i += 256) {
      const float dx = p[i * 3] - cx, dy = p[i * 3 + 1] - cy, dz = p[i * 3 + 2] - cz;
      const float d = dx * dx + dy * dy + dz * dz;
      if (d > lastd || (d == lastd && i > lasti)) {
        if (d < best || (d == best && i < besti)) { best = d; besti = i; }
      }
    }
    rd[t] = best; ri[t] = besti;
    __syncthreads();
    for (int s = 128; s > 0; s >>= 1) {
      if (t < s) {
        if (rd[t + s] < rd[t] || (rd[t + s] == rd[t] && ri[t + s] < ri[t])) {
          rd[t] = rd[t + s]; ri[t] = ri[t + s];
        }
      }
      __syncthreads();
    }
    if (t == 0) { knn_idx[bg * KN_ + j] = ri[0]; s_ld = rd[0]; s_li = ri[0]; }
    __syncthreads();
    lastd = s_ld; lasti = s_li;
    __syncthreads();
  }
}

__global__ void gather_feat_kernel(const float* __restrict__ feat,
                                   const int* __restrict__ kidx,
                                   float* __restrict__ grouped, long long n) {
  const long long i = (long long)blockIdx.x * blockDim.x + threadIdx.x;
  if (i >= n) return;
  const int c = (int)(i % C0_);
  const long long r = i / C0_;          // r = bg*KN_ + kk
  const int bg = (int)(r / KN_);
  const int b = bg / GC_;
  grouped[i] = feat[((long long)b * NPTS_ + kidx[r]) * C0_ + c];
}

// ---------------- reductions / elementwise ----------------
__global__ void groupmax_kernel(const float* __restrict__ x, float* __restrict__ y,
                                int Gn, int Kn, int C) {
  const long long i = (long long)blockIdx.x * blockDim.x + threadIdx.x;
  if (i >= (long long)Gn * C) return;
  const int g = (int)(i / C), c = (int)(i % C);
  float m = -1e30f;
  for (int k = 0; k < Kn; ++k) m = fmaxf(m, x[((long long)g * Kn + k) * C + c]);
  y[i] = m;
}

__global__ void bcast_add_kernel(float* __restrict__ h3, const float* __restrict__ t1,
                                 long long n, int Kn, int C) {
  const long long i = (long long)blockIdx.x * blockDim.x + threadIdx.x;
  if (i >= n) return;
  const long long row = i / C; const int c = (int)(i % C);
  h3[i] += t1[(row / Kn) * C + c];
}

__global__ void bn_stats_kernel(const float* __restrict__ x, int M, int C,
                                float* __restrict__ mean, float* __restrict__ rstd) {
  __shared__ float s1[256], s2[256];
  const int c = blockIdx.x, t = threadIdx.x;
  float a = 0.f, q = 0.f;
  for (int m = t; m < M; m += 256) {
    const float v = x[(long long)m * C + c];
    a += v; q += v * v;
  }
  s1[t] = a; s2[t] = q;
  __syncthreads();
  for (int s = 128; s > 0; s >>= 1) {
    if (t < s) { s1[t] += s1[t + s]; s2[t] += s2[t + s]; }
    __syncthreads();
  }
  if (t == 0) {
    const float mu = s1[0] / M;
    const float var = s2[0] / M - mu * mu;
    mean[c] = mu; rstd[c] = rsqrtf(var + EPS_);
  }
}

__global__ void bn_apply_kernel(float* __restrict__ x, const float* __restrict__ mean,
                                const float* __restrict__ rstd,
                                const float* __restrict__ g, const float* __restrict__ b,
                                long long n, int C, int relu) {
  const long long i = (long long)blockIdx.x * blockDim.x + threadIdx.x;
  if (i >= n) return;
  const int c = (int)(i % C);
  float v = (x[i] - mean[c]) * rstd[c] * g[c] + b[c];
  if (relu) v = fmaxf(v, 0.f);
  x[i] = v;
}

__global__ void layernorm_kernel(const float* __restrict__ x, const float* __restrict__ g,
                                 const float* __restrict__ b, float* __restrict__ y,
                                 int M, int C) {
  const int r = blockIdx.x * blockDim.x + threadIdx.x;
  if (r >= M) return;
  const float* xr = x + (long long)r * C;
  float m = 0.f;
  for (int c = 0; c < C; ++c) m += xr[c];
  m /= C;
  float v = 0.f;
  for (int c = 0; c < C; ++c) { const float d = xr[c] - m; v += d * d; }
  v /= C;
  const float rs = rsqrtf(v + EPS_);
  for (int c = 0; c < C; ++c) y[(long long)r * C + c] = (xr[c] - m) * rs * g[c] + b[c];
}

__global__ void silu_split_kernel(float* __restrict__ xs, const float* __restrict__ xz,
                                  long long n) {
  const long long i = (long long)blockIdx.x * blockDim.x + threadIdx.x;
  if (i >= n) return;
  const long long row = i / DI_; const int c = (int)(i % DI_);
  xs[i] = siluf_(xz[row * (2 * DI_) + c]);
}

__global__ void gate_kernel(float* __restrict__ ys, const float* __restrict__ xs,
                            const float* __restrict__ xz, const float* __restrict__ Dp,
                            long long n) {
  const long long i = (long long)blockIdx.x * blockDim.x + threadIdx.x;
  if (i >= n) return;
  const long long row = i / DI_; const int c = (int)(i % DI_);
  const float y = ys[i] + Dp[c] * xs[i];
  const float z = xz[row * (2 * DI_) + DI_ + c];
  ys[i] = y * siluf_(z);
}

// selective scan: one thread per (b, head, channel) chain; DSTATE=16 inner state.
__global__ void ssm_scan_kernel(const float* __restrict__ xs, const float* __restrict__ dt,
                                const float* __restrict__ proj, const float* __restrict__ A_log,
                                float* __restrict__ ys, int L, int reverse) {
  const int tid = blockIdx.x * blockDim.x + threadIdx.x;
  if (tid >= BN_ * DI_) return;
  const int b = tid / DI_;
  const int di = tid % DI_;
  const int hh = di / DH_;
  float Arow[DSTATE_], hstate[DSTATE_];
#pragma unroll
  for (int s = 0; s < DSTATE_; ++s) {
    Arow[s] = -__expf(A_log[di * DSTATE_ + s]);
    hstate[s] = 0.f;
  }
  const int boff = DTRANK_ + hh * DSTATE_;
  const int coff = DTRANK_ + HEADS_ * DSTATE_ + hh * DSTATE_;
  for (int step = 0; step < L; ++step) {
    const int l = reverse ? (L - 1 - step) : step;
    const long long row = (long long)b * L + l;
    const float dtv = dt[row * DI_ + di];
    const float xv  = xs[row * DI_ + di];
    const float* pr = proj + row * PROJW_;
    float y = 0.f;
#pragma unroll
    for (int s = 0; s < DSTATE_; ++s) {
      const float dA = __expf(dtv * Arow[s]);
      hstate[s] = dA * hstate[s] + dtv * xv * pr[boff + s];
      y += hstate[s] * pr[coff + s];
    }
    ys[row * DI_ + di] = y;
  }
}

// x_new = ssm_acc + gelu(dwconv3(h)) + x_old   (in-place on tok)
__global__ void dwconv_res_kernel(const float* __restrict__ h, const float* __restrict__ w,
                                  const float* __restrict__ bias,
                                  const float* __restrict__ ssm_acc, float* __restrict__ tok,
                                  int L, long long n) {
  const long long i = (long long)blockIdx.x * blockDim.x + threadIdx.x;
  if (i >= n) return;
  const int c = (int)(i % C1_);
  const long long bl = i / C1_;
  const int l = (int)(bl % L);
  const long long b = bl / L;
  float acc = bias[c];
#pragma unroll
  for (int j = 0; j < 3; ++j) {
    const int ll = l + j - 1;
    if (ll >= 0 && ll < L) acc += h[((long long)b * L + ll) * C1_ + c] * w[c * 3 + j];
  }
  tok[i] = ssm_acc[i] + geluf_(acc) + tok[i];
}

// ---------------- host orchestration ----------------
static inline int nblk(long long n) { return (int)((n + 255) / 256); }

extern "C" void kernel_launch(void* const* d_in, const int* in_sizes, int n_in,
                              void* d_out, int out_size, void* d_ws, size_t ws_size,
                              hipStream_t stream) {
  (void)in_sizes; (void)n_in; (void)out_size; (void)ws_size;
  auto F = [&](int i) -> const float* { return (const float*)d_in[i]; };

  // jax tree-flatten order: d_in[0]=xyz; params dict key-sorted.
  // block leaves (sorted): bwd{A_log,D,dt_b,dt_w,in_b,in_w,out_b,out_w,x_w},
  //   dw_b, dw_w, fwd{...same 9...}, ln1_b, ln1_g, ln2_b, ln2_g,
  //   mlp_b1, mlp_b2, mlp_be, mlp_g, mlp_w1, mlp_w2   -> 30 leaves/block
  const float* xyz = F(0);

  char* wp = (char*)d_ws;
  auto alloc = [&](size_t bytes) -> void* {
    void* p = (void*)wp;
    wp += (bytes + 255) & ~(size_t)255;
    return p;
  };
  const long long Me = (long long)BN_ * GC_ * KN_;   // 131072 encoder rows
  const int BL = BN_ * GC_;                          // 4096 tokens

  float* hid     = (float*)alloc((size_t)BN_ * NPTS_ * C0_ * 4);
  float* feat    = (float*)alloc((size_t)BN_ * NPTS_ * C0_ * 4);
  int*   fpsi    = (int*)  alloc((size_t)BN_ * GC_ * 4);
  float* center  = (float*)alloc((size_t)BN_ * GC_ * 3 * 4);
  int*   knni    = (int*)  alloc((size_t)BN_ * GC_ * KN_ * 4);
  float* grouped = (float*)alloc((size_t)Me * C0_ * 4);
  float* h1      = (float*)alloc((size_t)Me * C1_ * 4);
  float* h2      = (float*)alloc((size_t)Me * C1_ * 4);
  float* fg      = (float*)alloc((size_t)BL * C1_ * 4);
  float* t1      = (float*)alloc((size_t)BL * C1_ * 4);
  float* tok     = (float*)alloc((size_t)BL * C1_ * 4);
  float* hbuf    = (float*)alloc((size_t)BL * C1_ * 4);
  float* xzb     = (float*)alloc((size_t)BL * 2 * DI_ * 4);
  float* xsb     = (float*)alloc((size_t)BL * DI_ * 4);
  float* projb   = (float*)alloc((size_t)BL * PROJW_ * 4);
  float* dtb     = (float*)alloc((size_t)BL * DI_ * 4);
  float* ysb     = (float*)alloc((size_t)BL * DI_ * 4);
  float* ssm_acc = (float*)alloc((size_t)BL * C1_ * 4);
  float* mlp_h   = (float*)alloc((size_t)BL * MLPH_ * 4);
  float* bn_mean = (float*)alloc(1024 * 4);
  float* bn_rstd = (float*)alloc(1024 * 4);
  float* pooled  = (float*)alloc((size_t)BN_ * C1_ * 4);

  auto gemm = [&](const float* A, int lda, const float* W, int ldw, const float* bias,
                  float* Cst, int ldc, long long M, int N, int K, int act, int accum) {
    if ((M % 128 == 0) && (N % 32 == 0) && (K % 32 == 0) &&
        (lda % 4 == 0) && (ldw % 4 == 0)) {
      dim3 g((unsigned)(M / 128), (unsigned)(N / 32));
      gemm_f16_wmma_fast<<<g, dim3(256), 0, stream>>>(A, lda, W, ldw, bias, Cst, ldc,
                                                      (int)M, N, K, act, accum);
    } else {
      dim3 g((unsigned)((M + 31) / 32), (unsigned)((N + 63) / 64));
      gemm_f16_wmma<<<g, dim3(256), 0, stream>>>(A, lda, W, ldw, bias, Cst, ldc,
                                                 (int)M, N, K, act, accum);
    }
  };

  // ---- 1. point embedding: relu(xyz@W1+b1)@W2+b2 ----
  gemm(xyz, 3, F(75), C0_, F(73), hid, C0_, (long long)BN_ * NPTS_, C0_, 3, ACT_RELU, 0);
  gemm(hid, C0_, F(76), C0_, F(74), feat, C0_, (long long)BN_ * NPTS_, C0_, C0_, ACT_NONE, 0);

  // ---- 2. FPS + center gather ----
  fps_kernel<<<BN_, 256, 0, stream>>>(xyz, fpsi);
  gather_center_kernel<<<nblk(BN_ * GC_), 256, 0, stream>>>(xyz, fpsi, center);

  // ---- 3. KNN ----
  knn_kernel<<<BN_ * GC_, 256, 0, stream>>>(xyz, center, knni);

  // ---- 4. group gather ----
  gather_feat_kernel<<<nblk(Me * C0_), 256, 0, stream>>>(feat, knni, grouped, Me * C0_);

  // ---- 5. encoder ----
  gemm(grouped, C0_, F(65), C1_, F(61), h1, C1_, Me, C1_, C0_, ACT_NONE, 0);
  bn_stats_kernel<<<C1_, 256, 0, stream>>>(h1, (int)Me, C1_, bn_mean, bn_rstd);
  bn_apply_kernel<<<nblk(Me * C1_), 256, 0, stream>>>(h1, bn_mean, bn_rstd, F(64), F(63),
                                                      Me * C1_, C1_, 1);
  gemm(h1, C1_, F(66), C1_, F(62), h2, C1_, Me, C1_, C1_, ACT_NONE, 0);
  groupmax_kernel<<<nblk((long long)BL * C1_), 256, 0, stream>>>(h2, fg, BL, KN_, C1_);
  // concat([fg_bcast, f]) @ e2_w1 = fg@W_top + f@W_bot
  gemm(fg, C1_, F(71), C1_, F(67), t1, C1_, BL, C1_, C1_, ACT_NONE, 0);
  gemm(h2, C1_, F(71) + (size_t)C1_ * C1_, C1_, nullptr, h1, C1_, Me, C1_, C1_, ACT_NONE, 0);
  bcast_add_kernel<<<nblk(Me * C1_), 256, 0, stream>>>(h1, t1, Me * C1_, KN_, C1_);
  bn_stats_kernel<<<C1_, 256, 0, stream>>>(h1, (int)Me, C1_, bn_mean, bn_rstd);
  bn_apply_kernel<<<nblk(Me * C1_), 256, 0, stream>>>(h1, bn_mean, bn_rstd, F(70), F(69),
                                                      Me * C1_, C1_, 1);
  gemm(h1, C1_, F(72), C1_, F(68), h2, C1_, Me, C1_, C1_, ACT_NONE, 0);
  groupmax_kernel<<<nblk((long long)BL * C1_), 256, 0, stream>>>(h2, tok, BL, KN_, C1_);

  // ---- 6. HydraMamba blocks ----
  for (int blk = 0; blk < 2; ++blk) {
    const int pb = 1 + 30 * blk;
    layernorm_kernel<<<nblk(BL), 256, 0, stream>>>(tok, F(pb + 21), F(pb + 20), hbuf, BL, C1_);

    // bidirectional SSM: per-row ops commute with flip; only scan direction differs.
    auto run_ssm = [&](int base, int reverse, int accum) {
      const float* A_log = F(base + 0); const float* Dp   = F(base + 1);
      const float* dt_b  = F(base + 2); const float* dt_w = F(base + 3);
      const float* in_b  = F(base + 4); const float* in_w = F(base + 5);
      const float* out_b = F(base + 6); const float* out_w = F(base + 7);
      const float* x_w   = F(base + 8);
      gemm(hbuf, C1_, in_w, 2 * DI_, in_b, xzb, 2 * DI_, BL, 2 * DI_, C1_, ACT_NONE, 0);
      silu_split_kernel<<<nblk((long long)BL * DI_), 256, 0, stream>>>(xsb, xzb,
                                                                       (long long)BL * DI_);
      gemm(xsb, DI_, x_w, PROJW_, nullptr, projb, PROJW_, BL, PROJW_, DI_, ACT_NONE, 0);
      gemm(projb, PROJW_, dt_w, DI_, dt_b, dtb, DI_, BL, DI_, DTRANK_, ACT_SOFTPLUS, 0);
      ssm_scan_kernel<<<nblk(BN_ * DI_), 256, 0, stream>>>(xsb, dtb, projb, A_log, ysb,
                                                           GC_, reverse);
      gate_kernel<<<nblk((long long)BL * DI_), 256, 0, stream>>>(ysb, xsb, xzb, Dp,
                                                                 (long long)BL * DI_);
      gemm(ysb, DI_, out_w, C1_, out_b, ssm_acc, C1_, BL, C1_, DI_, ACT_NONE, accum);
    };
    run_ssm(pb + 11, /*reverse=*/0, /*accum=*/0);   // fwd
    run_ssm(pb + 0,  /*reverse=*/1, /*accum=*/1);   // bwd (outer flip cancels)

    dwconv_res_kernel<<<nblk((long long)BL * C1_), 256, 0, stream>>>(
        hbuf, F(pb + 10), F(pb + 9), ssm_acc, tok, GC_, (long long)BL * C1_);

    layernorm_kernel<<<nblk(BL), 256, 0, stream>>>(tok, F(pb + 23), F(pb + 22), hbuf, BL, C1_);
    gemm(hbuf, C1_, F(pb + 28), MLPH_, F(pb + 24), mlp_h, MLPH_, BL, MLPH_, C1_, ACT_NONE, 0);
    bn_stats_kernel<<<MLPH_, 256, 0, stream>>>(mlp_h, BL, MLPH_, bn_mean, bn_rstd);
    bn_apply_kernel<<<nblk((long long)BL * MLPH_), 256, 0, stream>>>(
        mlp_h, bn_mean, bn_rstd, F(pb + 27), F(pb + 26), (long long)BL * MLPH_, MLPH_, 1);
    gemm(mlp_h, MLPH_, F(pb + 29), C1_, F(pb + 25), tok, C1_, BL, C1_, MLPH_, ACT_NONE, 1);
  }

  // ---- 7. pool + head ----
  groupmax_kernel<<<nblk((long long)BN_ * C1_), 256, 0, stream>>>(tok, pooled, BN_, GC_, C1_);
  gemm(pooled, C1_, F(78), NCLS_, F(77), (float*)d_out, NCLS_, BN_, NCLS_, C1_, ACT_NONE, 0);
}
